// BackwardDecoder_2637109920362
// MI455X (gfx1250) — matrix operand order repaired and checked by reading, not compiled, not run on hardware
//
#include <hip/hip_runtime.h>

// ---------------- problem constants ----------------
#define Bsz   64
#define Ssz   40
#define Tsz   40
#define Hd    1024
#define Ed    512
#define Ad    1024
#define HEd   1024
#define OUTd  512
#define Vd    32000
#define H3    3072
#define EOS_ID 3
#define TCH   8            // teacher-forcing chunk (time steps)
#define RC    (TCH*Bsz)    // rows per TF chunk = 512

typedef __attribute__((ext_vector_type(16))) __bf16 v16bf;
typedef __attribute__((ext_vector_type(8)))  float  v8f;

#define GF_BIAS  1
#define GF_ACCUM 2
#define GF_BCAST 4

static inline int cdiv(int a, int b) { return (a + b - 1) / b; }

// ---------------------------------------------------------------------------
// Pack a KxN f32 weight matrix into the per-lane bf16 WMMA B-operand layout:
// tile (32k x 16n); lane = n%16 + 16*((k%32)/16); elem = k%16.
// Each lane's 16 bf16 values become one contiguous 32B chunk -> b128 loads,
// and the 4 N-tiles of a 64-col supertile are contiguous (+512 elems apart).
// ---------------------------------------------------------------------------
__global__ void pack_w_kernel(const float* __restrict__ src,
                              __bf16* __restrict__ dst, int K, int N) {
  size_t i = (size_t)blockIdx.x * blockDim.x + threadIdx.x;
  size_t total = (size_t)K * N;
  if (i >= total) return;
  int k = (int)(i / N);
  int n = (int)(i - (size_t)k * N);
  int kt = k >> 5, kin = k & 31;
  int nt = n >> 4, nin = n & 15;
  int lane = nin + ((kin >> 4) << 4);
  int elem = kin & 15;
  size_t o = (((size_t)kt * (N >> 4) + nt) * 32 + lane) * 16 + elem;
  dst[o] = (__bf16)src[i];
}

// ---------------------------------------------------------------------------
// GEMM: C(MxN,f32) = A(MxK,f32 -> bf16 in regs) * Bp(packed bf16)
//       [+ bias[n]] [+ bcast_src[(m%bmod)*ldc+n]] [+ C]   (flag-controlled)
// One wave computes a 16x64 supertile: A converted once per K-step,
// 4x v_wmma_f32_16x16x32_bf16 on contiguous packed-B tiles.
// Requires: M % 16 == 0, N % 64 == 0, K % 32 == 0 (all true here).
// ---------------------------------------------------------------------------
__global__ void __launch_bounds__(256)
gemm_bf16_kernel(const float* __restrict__ A, int lda,
                 const __bf16* __restrict__ Bp,
                 float* __restrict__ C, int ldc,
                 const float* __restrict__ bias,
                 const float* __restrict__ bsrc, int bmod,
                 int M, int N, int K, int flags) {
  const int lane = threadIdx.x & 31;
  const int wid  = blockIdx.x * (blockDim.x >> 5) + (threadIdx.x >> 5);
  const int ntn  = N >> 4;                  // 16-col tiles per row
  const int ntn4 = N >> 6;                  // 64-col supertiles per row
  const int tiles = (M >> 4) * ntn4;
  if (wid >= tiles) return;                 // wave-uniform exit (EXEC all-1)
  const int mt  = wid / ntn4;
  const int nt4 = wid - mt * ntn4;
  const int nt0 = nt4 << 2;                 // first 16-col tile of supertile
  const int m0  = mt << 4;
  const int hi  = lane >> 4;                // half-wave selector
  const int lm  = lane & 15;

  v8f acc[4] = {};
  const int ksteps = K >> 5;
  const float*  ap0 = A  + (size_t)(m0 + lm) * lda + (hi << 4);
  const __bf16* bp0 = Bp + ((size_t)nt0 * 32 + lane) * 16;
  const size_t  bstride = (size_t)ntn * 512;    // packed elems per K-step

  for (int kt = 0; kt < ksteps; ++kt) {
    const float* ap = ap0 + (kt << 5);
    v16bf a;
#pragma unroll
    for (int j = 0; j < 8; ++j) a[j]     = (__bf16)ap[j];       // K 0..7 / 16..23
#pragma unroll
    for (int j = 0; j < 8; ++j) a[8 + j] = (__bf16)ap[8 + j];   // K 8..15 / 24..31
    const __bf16* bp = bp0 + (size_t)kt * bstride;
    v16bf b0 = *(const v16bf*)(bp);
    v16bf b1 = *(const v16bf*)(bp + 512);
    v16bf b2 = *(const v16bf*)(bp + 1024);
    v16bf b3 = *(const v16bf*)(bp + 1536);
    __builtin_prefetch(bp + bstride, 0, 1);  // speculative next K-slice of B
    acc[0] = __builtin_amdgcn_wmma_f32_16x16x32_bf16(false, a, false, b0,
                                                     (short)0, acc[0], false, false);
    acc[1] = __builtin_amdgcn_wmma_f32_16x16x32_bf16(false, a, false, b1,
                                                     (short)0, acc[1], false, false);
    acc[2] = __builtin_amdgcn_wmma_f32_16x16x32_bf16(false, a, false, b2,
                                                     (short)0, acc[2], false, false);
    acc[3] = __builtin_amdgcn_wmma_f32_16x16x32_bf16(false, a, false, b3,
                                                     (short)0, acc[3], false, false);
  }

#pragma unroll
  for (int u = 0; u < 4; ++u) {
    const int col = ((nt0 + u) << 4) + lm;
    const float bv = (flags & GF_BIAS) ? bias[col] : 0.0f;
#pragma unroll
    for (int i = 0; i < 8; ++i) {
      int row = m0 + (hi << 3) + i;
      float v = acc[u][i] + bv;
      float* cp = C + (size_t)row * ldc + col;
      if (flags & GF_BCAST) v += bsrc[(size_t)(row % bmod) * ldc + col];
      if (flags & GF_ACCUM) v += *cp;
      *cp = v;
    }
  }
}

// ---------------- elementwise / reduction kernels ----------------
__device__ __forceinline__ float sigmoidf_(float x) {
  return 1.0f / (1.0f + __expf(-x));
}

// GRU gate combine: out = m*((1-z)*h + z*hc) + (1-m)*h ; h/gh rows taken mod hmod
__global__ void gru_combine_kernel(const float* __restrict__ gx,
                                   const float* __restrict__ gh,
                                   const float* __restrict__ h,
                                   const float* __restrict__ mask,
                                   float* __restrict__ out, int M, int hmod) {
  size_t tid = (size_t)blockIdx.x * blockDim.x + threadIdx.x;
  int r = (int)(tid / Hd), j = (int)(tid % Hd);
  if (r >= M) return;
  int rh = r % hmod;
  const float* gxr = gx + (size_t)r  * H3;
  const float* ghr = gh + (size_t)rh * H3;
  float z  = sigmoidf_(gxr[j]          + ghr[j]);
  float rr = sigmoidf_(gxr[Hd + j]     + ghr[Hd + j]);
  float hc = tanhf(gxr[2 * Hd + j] + rr * ghr[2 * Hd + j]);
  float hv = h[(size_t)rh * Hd + j];
  float hn = (1.0f - z) * hv + z * hc;
  float m  = mask[r];
  out[(size_t)r * Hd + j] = m * hn + (1.0f - m) * hv;
}

// e[s,r] = dot(tanh(sa[r,:] + uh[s, r%B, :]), a1_w) + a1_b   (one wave per (s,r))
__global__ void attn_e_kernel(const float* __restrict__ sa,
                              const float* __restrict__ uh,
                              const float* __restrict__ a1w,
                              const float* __restrict__ a1b,
                              float* __restrict__ eo, int R) {
  int lane = threadIdx.x & 31;
  int w = blockIdx.x * (blockDim.x >> 5) + (threadIdx.x >> 5);
  if (w >= Ssz * R) return;
  int s = w / R, r = w - s * R;
  int b = r & (Bsz - 1);
  const float* sar = sa + (size_t)r * Ad;
  const float* uhr = uh + ((size_t)s * Bsz + b) * Ad;
  float sum = 0.0f;
  for (int a = lane; a < Ad; a += 32)
    sum += tanhf(sar[a] + uhr[a]) * a1w[a];
#pragma unroll
  for (int m = 16; m; m >>= 1) sum += __shfl_xor(sum, m, 32);
  if (lane == 0) eo[(size_t)s * R + r] = sum + a1b[0];
}

// masked softmax over S per column r
__global__ void softmax_kernel(const float* __restrict__ e,
                               const float* __restrict__ xs_mask,
                               float* __restrict__ alpha, int R) {
  int r = blockIdx.x * blockDim.x + threadIdx.x;
  if (r >= R) return;
  int b = r & (Bsz - 1);
  float mx = -1e30f;
  for (int s = 0; s < Ssz; ++s) mx = fmaxf(mx, e[(size_t)s * R + r]);
  float sum = 0.0f;
  for (int s = 0; s < Ssz; ++s) {
    float w = __expf(e[(size_t)s * R + r] - mx) * xs_mask[s * Bsz + b];
    alpha[(size_t)s * R + r] = w;
    sum += w;
  }
  float inv = 1.0f / sum;
  for (int s = 0; s < Ssz; ++s) alpha[(size_t)s * R + r] *= inv;
}

// att[r,h] = sum_s alpha[s,r] * xs_h[s, r%B, h]
__global__ void attn_av_kernel(const float* __restrict__ alpha,
                               const float* __restrict__ xs_h,
                               float* __restrict__ att, int R) {
  size_t tid = (size_t)blockIdx.x * blockDim.x + threadIdx.x;
  int r = (int)(tid / HEd), h = (int)(tid % HEd);
  if (r >= R) return;
  int b = r & (Bsz - 1);
  float acc = 0.0f;
  for (int s = 0; s < Ssz; ++s)
    acc += alpha[(size_t)s * R + r] * xs_h[((size_t)s * Bsz + b) * HEd + h];
  att[(size_t)r * HEd + h] = acc;
}

// pairwise max-pool (2*OUT -> OUT), optional row-mask multiply
__global__ void maxpool_kernel(const float* __restrict__ l3,
                               float* __restrict__ out,
                               const float* __restrict__ mask, int M) {
  size_t tid = (size_t)blockIdx.x * blockDim.x + threadIdx.x;
  int r = (int)(tid / OUTd), o = (int)(tid % OUTd);
  if (r >= M) return;
  float v = fmaxf(l3[(size_t)r * (2 * OUTd) + 2 * o],
                  l3[(size_t)r * (2 * OUTd) + 2 * o + 1]);
  if (mask) v *= mask[r];
  out[(size_t)r * OUTd + o] = v;
}

// argmax over V per batch row (first-max tie break, like jnp.argmax)
__global__ void argmax_kernel(const float* __restrict__ scores,
                              int* __restrict__ nxt) {
  __shared__ float sv[256];
  __shared__ int   si[256];
  int b = blockIdx.x;
  const float* row = scores + (size_t)b * Vd;
  float best = -1e30f; int bi = 0;
  for (int i = threadIdx.x; i < Vd; i += 256) {
    float v = row[i];
    if (v > best) { best = v; bi = i; }
  }
  sv[threadIdx.x] = best; si[threadIdx.x] = bi;
  __syncthreads();
  for (int st = 128; st; st >>= 1) {
    if ((int)threadIdx.x < st) {
      float v2 = sv[threadIdx.x + st]; int i2 = si[threadIdx.x + st];
      if (v2 > sv[threadIdx.x] ||
          (v2 == sv[threadIdx.x] && i2 < si[threadIdx.x])) {
        sv[threadIdx.x] = v2; si[threadIdx.x] = i2;
      }
    }
    __syncthreads();
  }
  if (threadIdx.x == 0) nxt[b] = si[0];
}

// embedding gather: ye[r,:] = emb[ids[r],:]
__global__ void gather_emb_kernel(const float* __restrict__ emb,
                                  const int* __restrict__ ids,
                                  float* __restrict__ ye, int rows) {
  size_t tid = (size_t)blockIdx.x * blockDim.x + threadIdx.x;
  int r = (int)(tid / Ed), e = (int)(tid % Ed);
  if (r >= rows) return;
  ye[(size_t)r * Ed + e] = emb[(size_t)ids[r] * Ed + e];
}

// greedy carry update: ye = emb[nxt]; ym' = (ym>0.5 && !mnt); mnt' = (nxt==EOS)
__global__ void greedy_update_kernel(const float* __restrict__ emb,
                                     const int* __restrict__ nxt,
                                     float* __restrict__ ye,
                                     float* __restrict__ ym,
                                     int* __restrict__ mnt) {
  size_t tid = (size_t)blockIdx.x * blockDim.x + threadIdx.x;
  int b = (int)(tid / Ed), e = (int)(tid % Ed);
  if (b >= Bsz) return;
  int id = nxt[b];
  ye[(size_t)b * Ed + e] = emb[(size_t)id * Ed + e];
  if (e == 0) {
    float m = ym[b]; int dead = mnt[b];
    ym[b]  = (m > 0.5f && !dead) ? 1.0f : 0.0f;
    mnt[b] = (id == EOS_ID) ? 1 : 0;
  }
}

// states[t] = s_t * ym ; m[t] = ym   (ym is the pre-update mask of this step)
__global__ void store_states_kernel(const float* __restrict__ s_t,
                                    const float* __restrict__ ym,
                                    float* __restrict__ out_states,
                                    float* __restrict__ out_m, int t) {
  size_t tid = (size_t)blockIdx.x * blockDim.x + threadIdx.x;
  int b = (int)(tid / Hd), h = (int)(tid % Hd);
  if (b >= Bsz) return;
  float m = ym[b];
  out_states[((size_t)t * Bsz + b) * Hd + h] = s_t[(size_t)b * Hd + h] * m;
  if (h == 0) out_m[(size_t)t * Bsz + b] = m;
}

__global__ void init_mask_kernel(float* ym, int* mnt) {
  int b = threadIdx.x;
  if (b < Bsz) { ym[b] = 1.0f; mnt[b] = 0; }
}

// ---------------- host orchestration ----------------
static void launch_gemm(const float* A, int lda, const __bf16* Bp,
                        float* C, int ldc, const float* bias,
                        const float* bsrc, int bmod,
                        int M, int N, int K, int flags, hipStream_t s) {
  int waves = (M / 16) * (N / 64);   // one wave per 16x64 supertile
  gemm_bf16_kernel<<<cdiv(waves, 8), 256, 0, s>>>(A, lda, Bp, C, ldc, bias,
                                                  bsrc, bmod, M, N, K, flags);
}
static void launch_pack(const float* src, __bf16* dst, int K, int N,
                        hipStream_t s) {
  size_t n = (size_t)K * N;
  pack_w_kernel<<<(int)((n + 255) / 256), 256, 0, s>>>(src, dst, K, N);
}
#define EW(total) <<<cdiv((int)(total), 256), 256, 0, stream>>>

extern "C" void kernel_launch(void* const* d_in, const int* in_sizes, int n_in,
                              void* d_out, int out_size, void* d_ws,
                              size_t ws_size, hipStream_t stream) {
  (void)in_sizes; (void)n_in; (void)out_size; (void)ws_size;
  // inputs (setup_inputs dict order)
  const float* s_tm1   = (const float*)d_in[0];
  const float* xs_h    = (const float*)d_in[1];
  const int*   ys      = (const int*)  d_in[2];
  const float* uh      = (const float*)d_in[3];
  const float* xs_mask = (const float*)d_in[4];
  const float* ys_mask = (const float*)d_in[5];
  const float* emb     = (const float*)d_in[6];
  const float* g1_wx = (const float*)d_in[7],  *g1_wh = (const float*)d_in[8];
  const float* g1_b  = (const float*)d_in[9];
  const float* g2_wx = (const float*)d_in[10], *g2_wh = (const float*)d_in[11];
  const float* g2_b  = (const float*)d_in[12];
  const float* sa_w  = (const float*)d_in[13], *sa_b  = (const float*)d_in[14];
  const float* a1_w  = (const float*)d_in[15], *a1_b  = (const float*)d_in[16];
  const float* ls_w  = (const float*)d_in[17], *ls_b  = (const float*)d_in[18];
  const float* ly_w  = (const float*)d_in[19], *ly_b  = (const float*)d_in[20];
  const float* lc_w  = (const float*)d_in[21], *lc_b  = (const float*)d_in[22];
  const float* cls_w = (const float*)d_in[23], *cls_b = (const float*)d_in[24];

  float* out      = (float*)d_out;
  float* out_logit  = out;                                   // (T,B,OUT)
  float* out_states = out + (size_t)Tsz * Bsz * OUTd;        // (41,B,H)
  float* out_m      = out_states + (size_t)41 * Bsz * Hd;    // (41,B)

  // ---- workspace carve-up ----
  char* wp = (char*)d_ws;
  auto alloc = [&](size_t bytes) -> void* {
    void* p = (void*)wp;
    wp += (bytes + 255) & ~(size_t)255;
    return p;
  };
  __bf16* Wg1x = (__bf16*)alloc((size_t)Ed  * H3 * 2);
  __bf16* Wg1h = (__bf16*)alloc((size_t)Hd  * H3 * 2);
  __bf16* Wg2x = (__bf16*)alloc((size_t)HEd * H3 * 2);
  __bf16* Wg2h = (__bf16*)alloc((size_t)Hd  * H3 * 2);
  __bf16* Wsa  = (__bf16*)alloc((size_t)Hd  * Ad * 2);
  __bf16* Wls  = (__bf16*)alloc((size_t)Hd  * 2 * OUTd * 2);
  __bf16* Wly  = (__bf16*)alloc((size_t)Ed  * 2 * OUTd * 2);
  __bf16* Wlc  = (__bf16*)alloc((size_t)HEd * 2 * OUTd * 2);
  __bf16* Wcls = (__bf16*)alloc((size_t)OUTd * Vd * 2);

  float* ye      = (float*)alloc((size_t)RC * Ed * 4);
  float* gx      = (float*)alloc((size_t)RC * H3 * 4);
  float* gh      = (float*)alloc((size_t)Bsz * H3 * 4);   // small / broadcast
  float* s_above = (float*)alloc((size_t)RC * Hd * 4);
  float* sa      = (float*)alloc((size_t)RC * Ad * 4);
  float* ebuf    = (float*)alloc((size_t)Ssz * RC * 4);
  float* alpha   = (float*)alloc((size_t)Ssz * RC * 4);
  float* att     = (float*)alloc((size_t)RC * HEd * 4);
  float* logit3  = (float*)alloc((size_t)RC * 2 * OUTd * 4);
  float* pooled  = (float*)alloc((size_t)Bsz * OUTd * 4);
  float* scores  = (float*)alloc((size_t)Bsz * Vd * 4);
  float* sA      = (float*)alloc((size_t)Bsz * Hd * 4);
  float* sB      = (float*)alloc((size_t)Bsz * Hd * 4);
  float* ls_sml  = (float*)alloc((size_t)Bsz * 2 * OUTd * 4);
  float* ym      = (float*)alloc(Bsz * 4);
  int*   mnt     = (int*)  alloc(Bsz * 4);
  int*   nxt     = (int*)  alloc(Bsz * 4);

  // ---- pack weights to bf16 WMMA layout (fits in 192MB L2 afterwards) ----
  launch_pack(g1_wx, Wg1x, Ed,  H3, stream);
  launch_pack(g1_wh, Wg1h, Hd,  H3, stream);
  launch_pack(g2_wx, Wg2x, HEd, H3, stream);
  launch_pack(g2_wh, Wg2h, Hd,  H3, stream);
  launch_pack(sa_w,  Wsa,  Hd,  Ad, stream);
  launch_pack(ls_w,  Wls,  Hd,  2 * OUTd, stream);
  launch_pack(ly_w,  Wly,  Ed,  2 * OUTd, stream);
  launch_pack(lc_w,  Wlc,  HEd, 2 * OUTd, stream);
  launch_pack(cls_w, Wcls, OUTd, Vd, stream);

  // =====================  greedy phase (41 sequential steps)  ==============
  init_mask_kernel<<<1, 64, 0, stream>>>(ym, mnt);
  gather_emb_kernel EW(Bsz * Ed) (emb, ys, ye, Bsz);   // ye = emb[ys[0]]
  const float* s_cur = s_tm1;
  float* bufs[2] = {sA, sB};
  int pp = 0;
  for (int t = 0; t <= Tsz; ++t) {
    float* s_nxt = bufs[pp];
    // GRU 1: s_above = gru(ye, s_cur)
    launch_gemm(ye, Ed, Wg1x, gx, H3, g1_b, nullptr, 0, Bsz, H3, Ed, GF_BIAS, stream);
    launch_gemm(s_cur, Hd, Wg1h, gh, H3, nullptr, nullptr, 0, Bsz, H3, Hd, 0, stream);
    gru_combine_kernel EW(Bsz * Hd) (gx, gh, s_cur, ym, s_above, Bsz, Bsz);
    // attention
    launch_gemm(s_above, Hd, Wsa, sa, Ad, sa_b, nullptr, 0, Bsz, Ad, Hd, GF_BIAS, stream);
    attn_e_kernel<<<cdiv(Ssz * Bsz, 8), 256, 0, stream>>>(sa, uh, a1_w, a1_b, ebuf, Bsz);
    softmax_kernel EW(Bsz) (ebuf, xs_mask, alpha, Bsz);
    attn_av_kernel EW(Bsz * HEd) (alpha, xs_h, att, Bsz);
    // GRU 2: s_t = gru(att, s_above)
    launch_gemm(att, HEd, Wg2x, gx, H3, g2_b, nullptr, 0, Bsz, H3, HEd, GF_BIAS, stream);
    launch_gemm(s_above, Hd, Wg2h, gh, H3, nullptr, nullptr, 0, Bsz, H3, Hd, 0, stream);
    gru_combine_kernel EW(Bsz * Hd) (gx, gh, s_above, ym, s_nxt, Bsz, Bsz);
    store_states_kernel EW(Bsz * Hd) (s_nxt, ym, out_states, out_m, t);
    if (t < Tsz) {  // last step's argmax feeds nothing
      // step_out -> scores -> argmax -> carry update
      launch_gemm(s_nxt, Hd, Wls, logit3, 2 * OUTd, ls_b, nullptr, 0, Bsz, 2 * OUTd, Hd, GF_BIAS, stream);
      launch_gemm(ye, Ed, Wly, logit3, 2 * OUTd, ly_b, nullptr, 0, Bsz, 2 * OUTd, Ed, GF_BIAS | GF_ACCUM, stream);
      launch_gemm(att, HEd, Wlc, logit3, 2 * OUTd, lc_b, nullptr, 0, Bsz, 2 * OUTd, HEd, GF_BIAS | GF_ACCUM, stream);
      maxpool_kernel EW(Bsz * OUTd) (logit3, pooled, nullptr, Bsz);
      launch_gemm(pooled, OUTd, Wcls, scores, Vd, cls_b, nullptr, 0, Bsz, Vd, OUTd, GF_BIAS, stream);
      argmax_kernel<<<Bsz, 256, 0, stream>>>(scores, nxt);
      greedy_update_kernel EW(Bsz * Ed) (emb, nxt, ye, ym, mnt);
    }
    s_cur = s_nxt;
    pp ^= 1;
  }

  // =============  teacher-forcing phase (independent steps, chunked)  ======
  // gh = s_tm1 @ g1_wh  (broadcast across all t); ls_sml = s_tm1@ls_w + ls_b
  launch_gemm(s_tm1, Hd, Wg1h, gh, H3, nullptr, nullptr, 0, Bsz, H3, Hd, 0, stream);
  launch_gemm(s_tm1, Hd, Wls, ls_sml, 2 * OUTd, ls_b, nullptr, 0, Bsz, 2 * OUTd, Hd, GF_BIAS, stream);
  for (int t0 = 0; t0 < Tsz; t0 += TCH) {
    const int rows = RC;
    gather_emb_kernel EW(rows * Ed) (emb, ys + (size_t)t0 * Bsz, ye, rows);
    launch_gemm(ye, Ed, Wg1x, gx, H3, g1_b, nullptr, 0, rows, H3, Ed, GF_BIAS, stream);
    gru_combine_kernel EW(rows * Hd) (gx, gh, s_tm1, ys_mask + (size_t)t0 * Bsz,
                                      s_above, rows, Bsz);
    launch_gemm(s_above, Hd, Wsa, sa, Ad, sa_b, nullptr, 0, rows, Ad, Hd, GF_BIAS, stream);
    attn_e_kernel<<<cdiv(Ssz * rows, 8), 256, 0, stream>>>(sa, uh, a1_w, a1_b, ebuf, rows);
    softmax_kernel EW(rows) (ebuf, xs_mask, alpha, rows);
    attn_av_kernel EW(rows * HEd) (alpha, xs_h, att, rows);
    // logit3 = att@lc_w + lc_b + bcast(ls_sml) ; += ye@ly_w + ly_b
    launch_gemm(att, HEd, Wlc, logit3, 2 * OUTd, lc_b, ls_sml, Bsz,
                rows, 2 * OUTd, HEd, GF_BIAS | GF_BCAST, stream);
    launch_gemm(ye, Ed, Wly, logit3, 2 * OUTd, ly_b, nullptr, 0,
                rows, 2 * OUTd, Ed, GF_BIAS | GF_ACCUM, stream);
    maxpool_kernel EW(rows * OUTd) (logit3, out_logit + (size_t)t0 * Bsz * OUTd,
                                    ys_mask + (size_t)t0 * Bsz, rows);
  }
}
